// NBEATSSeasonalBlockKAN_53652731461734
// MI455X (gfx1250) — compile-verified, matrix operand order checked by reading, not run on hardware
//
#include <hip/hip_runtime.h>
#include <hip/hip_bf16.h>
#include <stdint.h>

// ---------------- problem constants (from reference) ----------------
#define BSZ   2048
#define BACKC 512
#define UNITS 512
#define FOREC 128
#define NLAY  4
#define NCOEF 8                      // NUM + K = 5 + 3
#define KDIM  (UNITS + UNITS*NCOEF)  // 4608 : [silu | bspline-flat]

// WMMA types (gfx1250, wave32)
typedef __attribute__((ext_vector_type(16))) __bf16 v16bf;
typedef __attribute__((ext_vector_type(8)))  float  v8f;

union Pack8  { __bf16 h[8]; uint4 u; };
union FragBf { v16bf v; uint4 u[2]; };

// ---------------- gfx1250 async global->LDS copy helpers ----------------
// Generic pointers to __shared__ are {LDS_aperture_hi32, lds_offset}; the
// async instruction's VDST wants the wave-relative LDS byte offset = low 32b.
__device__ __forceinline__ uint32_t lds_off(const void* p) {
    return (uint32_t)(uint64_t)p;
}
// GLOBAL_LOAD_ASYNC_TO_LDS_B128 (GV mode): per-lane 16B global->LDS DMA,
// tracked by ASYNCcnt (CDNA5 ISA).
__device__ __forceinline__ void async_ld_b128(uint32_t lds, const void* g) {
    asm volatile("global_load_async_to_lds_b128 %0, %1, off"
                 :: "v"(lds), "v"(g) : "memory");
}
__device__ __forceinline__ void wait_async0() {
    asm volatile("s_wait_asynccnt 0x0" ::: "memory");
}

// ---------------- weight packing: Bt layout = N x K (K contiguous) ----------
// Bt[o*KDIM + row];  row < UNITS -> sb[row,o];  row = UNITS+i*8+k -> coef*ssp
__global__ __launch_bounds__(256) void pack_w_fc(
    const float* __restrict__ coef, const float* __restrict__ sb,
    const float* __restrict__ ssp, __bf16* __restrict__ Wt)
{
    size_t idx = (size_t)blockIdx.x * 256 + threadIdx.x;
    const size_t total = (size_t)NLAY * UNITS * KDIM;
    if (idx >= total) return;
    int l   = (int)(idx / ((size_t)UNITS * KDIM));
    int rem = (int)(idx % ((size_t)UNITS * KDIM));
    int o   = rem / KDIM;
    int row = rem % KDIM;
    float w;
    if (row < UNITS) {
        w = sb[((size_t)l * UNITS + row) * UNITS + o];
    } else {
        int t = row - UNITS;
        int i = t >> 3, k = t & 7;
        size_t io = ((size_t)l * UNITS + i) * UNITS + o;
        w = coef[io * NCOEF + k] * ssp[io];
    }
    Wt[idx] = (__bf16)w;
}

__global__ __launch_bounds__(256) void pack_w_th(
    const float* __restrict__ coef, const float* __restrict__ sb,
    const float* __restrict__ ssp, __bf16* __restrict__ Wt)
{
    size_t idx = (size_t)blockIdx.x * 256 + threadIdx.x;
    const size_t total = (size_t)FOREC * KDIM;
    if (idx >= total) return;
    int o   = (int)(idx / KDIM);
    int row = (int)(idx % KDIM);
    float w;
    if (row < UNITS) {
        w = sb[(size_t)row * FOREC + o];
    } else {
        int t = row - UNITS;
        int i = t >> 3, k = t & 7;
        size_t io = (size_t)i * FOREC + o;
        w = coef[io * NCOEF + k] * ssp[io];
    }
    Wt[idx] = (__bf16)w;
}

// ---------------- seasonal bases, transposed: St[n*128 + k] ----------------
__global__ __launch_bounds__(256) void seasonal_basis(
    __bf16* __restrict__ Sbt, __bf16* __restrict__ Sft)
{
    int idx = blockIdx.x * 256 + threadIdx.x;           // 640 * 128
    if (idx >= (BACKC + FOREC) * 128) return;
    int t = idx / 128;                                  // time index 0..639
    int r = idx % 128;                                  // harmonic index
    const float T = (float)(BACKC + FOREC);             // 640
    float freq = (float)(r & 63) * (T / 63.0f);         // linspace(0, T, 64)
    float arg  = 6.283185307179586f * freq * ((float)t / T);
    float v    = (r >= 64) ? __sinf(arg) : __cosf(arg);
    if (t < BACKC) Sbt[(size_t)t * 128 + r]           = (__bf16)v;
    else           Sft[(size_t)(t - BACKC) * 128 + r] = (__bf16)v;
}

// ---------------- KAN activation prep: silu + cubic B-spline basis ----------
// grid points t_j = -1 + (j-3)*0.4, j = 0..11 (uniform, spacing 0.4)
__global__ __launch_bounds__(256) void kan_prep(
    const float* __restrict__ h, __bf16* __restrict__ A)
{
    int idx = blockIdx.x * 256 + threadIdx.x;           // BSZ * UNITS
    if (idx >= BSZ * UNITS) return;
    int b = idx / UNITS;
    int i = idx % UNITS;
    float xv = h[idx];

    float s = xv / (1.0f + __expf(-xv));                // silu
    A[(size_t)b * KDIM + i] = (__bf16)s;

    float v[11];
#pragma unroll
    for (int j = 0; j < 11; ++j) {
        float t0 = -2.2f + 0.4f * j;
        v[j] = (xv >= t0 && xv < t0 + 0.4f) ? 1.0f : 0.0f;
    }
#pragma unroll
    for (int p = 1; p <= 3; ++p) {
        float inv = 1.0f / (0.4f * (float)p);
#pragma unroll 11
        for (int j = 0; j < 11 - p; ++j) {
            float tj = -2.2f + 0.4f * j;
            v[j] = (xv - tj) * inv * v[j] + ((tj + 0.4f * (p + 1)) - xv) * inv * v[j + 1];
        }
    }
    Pack8 pk;
#pragma unroll
    for (int k = 0; k < 8; ++k) pk.h[k] = (__bf16)v[k];
    *(uint4*)(A + (size_t)b * KDIM + UNITS + i * 8) = pk.u;
}

__global__ __launch_bounds__(256) void cvt_f32_bf16(
    const float* __restrict__ in, __bf16* __restrict__ out, int n)
{
    int idx = blockIdx.x * 256 + threadIdx.x;
    if (idx < n) out[idx] = (__bf16)in[idx];
}

// ---------------- tiled WMMA bf16 GEMM: C(MxN f32) = A(MxK) @ Bt(NxK)^T ------
// 128x128 tile / workgroup, 8 wave32 waves (4M x 2N), each wave 32x64 =
// 2x4 v_wmma_f32_16x16x32_bf16 accumulators. Double-buffered LDS staged by
// async global->LDS DMA, prefetch of tile k+1 overlapped with WMMA of tile k.
#define TM 128
#define TN 128
#define TK 64

__global__ __launch_bounds__(256) void gemm_bf16_wmma(
    const __bf16* __restrict__ A, const __bf16* __restrict__ Bt,
    float* __restrict__ C, int M, int N, int Kd)      // Kd/TK must be even
{
    __shared__ __align__(16) __bf16 sA[2][TM * TK];   // 2 x 16KB
    __shared__ __align__(16) __bf16 sB[2][TN * TK];   // 2 x 16KB

    const int tid   = threadIdx.x;
    const int lane  = tid & 31;
    const int wave  = tid >> 5;
    const int half  = lane >> 4;        // 0: K 0..7/16..23 ; 1: K 8..15/24..31
    const int lidx  = lane & 15;
    const int waveM = (wave & 3) * 32;
    const int waveN = (wave >> 2) * 64;
    const int blkM  = blockIdx.y * TM;
    const int blkN  = blockIdx.x * TN;

    // staging chunk map: 1024 16B-chunks per tile, 4 per thread (TM==TN)
    const __bf16* gA[4];
    const __bf16* gB[4];
    uint32_t dA[2][4], dB[2][4];
#pragma unroll
    for (int j = 0; j < 4; ++j) {
        int c   = tid + j * 256;        // 0..1023
        int row = c >> 3;               // 0..127 (m for A, n for B)
        int cc  = (c & 7) * 8;          // k-offset within TK
        gA[j] = A  + (size_t)(blkM + row) * Kd + cc;
        gB[j] = Bt + (size_t)(blkN + row) * Kd + cc;
#pragma unroll
        for (int s = 0; s < 2; ++s) {
            dA[s][j] = lds_off(&sA[s][row * TK + cc]);
            dB[s][j] = lds_off(&sB[s][row * TK + cc]);
        }
    }

    v8f acc[2][4];
#pragma unroll
    for (int mi = 0; mi < 2; ++mi)
#pragma unroll
        for (int ni = 0; ni < 4; ++ni) acc[mi][ni] = (v8f)0.0f;

    const int nIter = Kd / TK;

    // prologue: issue tile 0 into buffer 0
#pragma unroll
    for (int j = 0; j < 4; ++j) {
        async_ld_b128(dA[0][j], gA[j]);
        async_ld_b128(dB[0][j], gB[j]);
    }

    auto do_step = [&](int bsel, int it) {
        wait_async0();                  // this wave's DMA into sX[bsel] done
        __syncthreads();                // everyone's DMA done
        if (it + 1 < nIter) {           // prefetch next tile into other buffer
            size_t koff = (size_t)(it + 1) * TK;
#pragma unroll
            for (int j = 0; j < 4; ++j) {
                async_ld_b128(dA[bsel ^ 1][j], gA[j] + koff);
                async_ld_b128(dB[bsel ^ 1][j], gB[j] + koff);
            }
        }
        const __bf16* bAp = sA[bsel];
        const __bf16* bBp = sB[bsel];
#pragma unroll
        for (int kk = 0; kk < TK; kk += 32) {
            FragBf aF[2], bF[4];
#pragma unroll
            for (int mi = 0; mi < 2; ++mi) {
                const __bf16* p = bAp + (waveM + mi * 16 + lidx) * TK + kk + half * 8;
                aF[mi].u[0] = *(const uint4*)p;          // K +0..7
                aF[mi].u[1] = *(const uint4*)(p + 16);   // K +16..23
            }
#pragma unroll
            for (int ni = 0; ni < 4; ++ni) {
                const __bf16* p = bBp + (waveN + ni * 16 + lidx) * TK + kk + half * 8;
                bF[ni].u[0] = *(const uint4*)p;
                bF[ni].u[1] = *(const uint4*)(p + 16);
            }
#pragma unroll
            for (int mi = 0; mi < 2; ++mi)
#pragma unroll
                for (int ni = 0; ni < 4; ++ni)
                    acc[mi][ni] = __builtin_amdgcn_wmma_f32_16x16x32_bf16(
                        false, aF[mi].v, false, bF[ni].v,
                        (short)0, acc[mi][ni], false, false);
        }
        __syncthreads();                // all reads of sX[bsel] complete
    };

    for (int it = 0; it < nIter; it += 2) {   // nIter even for all call sites
        do_step(0, it);
        do_step(1, it + 1);
    }

    // epilogue: v8f element r -> row = r + 8*half, col = lidx
#pragma unroll
    for (int mi = 0; mi < 2; ++mi)
#pragma unroll
        for (int ni = 0; ni < 4; ++ni)
#pragma unroll
            for (int r = 0; r < 8; ++r) {
                int row = blkM + waveM + mi * 16 + r + 8 * half;
                int col = blkN + waveN + ni * 16 + lidx;
                C[(size_t)row * N + col] = acc[mi][ni][r];
            }
}

// ---------------- host-side launch ----------------
extern "C" void kernel_launch(void* const* d_in, const int* in_sizes, int n_in,
                              void* d_out, int out_size, void* d_ws, size_t ws_size,
                              hipStream_t stream) {
    const float* x       = (const float*)d_in[0];  // (B, 512)
    const float* coef_fc = (const float*)d_in[1];  // (4, 512, 512, 8)
    const float* sb_fc   = (const float*)d_in[2];  // (4, 512, 512)
    const float* ssp_fc  = (const float*)d_in[3];  // (4, 512, 512)
    const float* coef_th = (const float*)d_in[4];  // (512, 128, 8)
    const float* sb_th   = (const float*)d_in[5];  // (512, 128)
    const float* ssp_th  = (const float*)d_in[6];  // (512, 128)
    float* out = (float*)d_out;                    // backcast (B,512) ++ forecast (B,128)

    // workspace carve (all chunks 16B-multiple)
    char* w = (char*)d_ws;
    __bf16* Abuf  = (__bf16*)w; w += (size_t)BSZ * KDIM * 2;          // 18.9 MB
    __bf16* Wfc   = (__bf16*)w; w += (size_t)NLAY * UNITS * KDIM * 2; // 18.9 MB (N x K)
    __bf16* Wth   = (__bf16*)w; w += (size_t)FOREC * KDIM * 2;        //  1.2 MB (N x K)
    float*  hbuf  = (float*) w; w += (size_t)BSZ * UNITS * 4;         //  4.2 MB
    float*  theta = (float*) w; w += (size_t)BSZ * FOREC * 4;         //  1.0 MB
    __bf16* thb   = (__bf16*)w; w += (size_t)BSZ * FOREC * 2;         //  0.5 MB
    __bf16* Sbt   = (__bf16*)w; w += (size_t)BACKC * 128 * 2;         //  (N x K)
    __bf16* Sft   = (__bf16*)w; w += (size_t)FOREC * 128 * 2;         //  (N x K)

    // 1) pack weights + bases (independent)
    {
        size_t tot = (size_t)NLAY * UNITS * KDIM;
        pack_w_fc<<<dim3((unsigned)((tot + 255) / 256)), 256, 0, stream>>>(coef_fc, sb_fc, ssp_fc, Wfc);
    }
    {
        size_t tot = (size_t)FOREC * KDIM;
        pack_w_th<<<dim3((unsigned)((tot + 255) / 256)), 256, 0, stream>>>(coef_th, sb_th, ssp_th, Wth);
    }
    seasonal_basis<<<dim3((128 * (BACKC + FOREC) + 255) / 256), 256, 0, stream>>>(Sbt, Sft);

    // 2) 4 KAN fc layers: prep (silu+bspline -> bf16 A) then WMMA GEMM
    const float* cur = x;
    for (int l = 0; l < NLAY; ++l) {
        kan_prep<<<dim3(BSZ * UNITS / 256), 256, 0, stream>>>(cur, Abuf);
        gemm_bf16_wmma<<<dim3(UNITS / TN, BSZ / TM), 256, 0, stream>>>(
            Abuf, Wfc + (size_t)l * UNITS * KDIM, hbuf, BSZ, UNITS, KDIM);
        cur = hbuf;
    }

    // 3) KAN head -> theta (B x 128)   (nIter = 72, even)
    kan_prep<<<dim3(BSZ * UNITS / 256), 256, 0, stream>>>(hbuf, Abuf);
    gemm_bf16_wmma<<<dim3(FOREC / TN, BSZ / TM), 256, 0, stream>>>(
        Abuf, Wth, theta, BSZ, FOREC, KDIM);

    // 4) theta @ Sb / theta @ Sf  (K = 128 -> nIter = 2, even)
    cvt_f32_bf16<<<dim3(BSZ * FOREC / 256), 256, 0, stream>>>(theta, thb, BSZ * FOREC);
    gemm_bf16_wmma<<<dim3(BACKC / TN, BSZ / TM), 256, 0, stream>>>(
        thb, Sbt, out, BSZ, BACKC, FOREC);
    gemm_bf16_wmma<<<dim3(FOREC / TN, BSZ / TM), 256, 0, stream>>>(
        thb, Sft, out + (size_t)BSZ * BACKC, BSZ, FOREC, FOREC);
}
